// GAT_7086696038725
// MI455X (gfx1250) — matrix operand order repaired and checked by reading, not compiled, not run on hardware
//
#include <hip/hip_runtime.h>
#include <hip/hip_bf16.h>

#define GAT_N 50000
#define GAT_E 800000
#define NEG_SLOPE 0.2f

typedef __attribute__((ext_vector_type(16))) __bf16 v16bf;
typedef __attribute__((ext_vector_type(8)))  __bf16 v8bf;
typedef __attribute__((ext_vector_type(4)))  __bf16 v4bf;
typedef __attribute__((ext_vector_type(8)))  float  v8f;

// ---------------------------------------------------------------------------
// f32 -> bf16 row-major conversion (4 elements/thread, b128 in / b64 out).
// ---------------------------------------------------------------------------
__global__ void gat_cvt_bf16(const float* __restrict__ in, __bf16* __restrict__ out,
                             int n4) {
  int i = blockIdx.x * blockDim.x + threadIdx.x;
  if (i >= n4) return;
  float4 v = ((const float4*)in)[i];
  v4bf o;
  o[0] = (__bf16)v.x; o[1] = (__bf16)v.y; o[2] = (__bf16)v.z; o[3] = (__bf16)v.w;
  ((v4bf*)out)[i] = o;
}

// ---------------------------------------------------------------------------
// Pack W[K,N] (f32, row-major) into the WMMA B-fragment layout, bf16:
//   flat = ((kt*tilesN + tn)*32 + lane)*16 + e
//   element(lane,e) = W[kt*32 + (lane>>4)*16 + e][tn*16 + (lane&15)]
// so a wave's B fragment is 16 contiguous bf16 per lane (two b128 loads).
// ---------------------------------------------------------------------------
__global__ void gat_pack_w(const float* __restrict__ W, __bf16* __restrict__ out,
                           int K, int N) {
  int i = blockIdx.x * blockDim.x + threadIdx.x;
  if (i >= K * N) return;
  int e    = i & 15;
  int lane = (i >> 4) & 31;
  int rest = i >> 9;
  int tilesN = N >> 4;
  int tn = rest % tilesN;
  int kt = rest / tilesN;
  int hl = lane >> 4, l = lane & 15;
  out[i] = (__bf16)W[(size_t)(kt * 32 + hl * 16 + e) * N + tn * 16 + l];
}

// ---------------------------------------------------------------------------
// WMMA GEMM: C[M,N] = A[M,K] @ B[K,N], A bf16 row-major, B pre-packed bf16,
// C f32. One wave computes a 16x64 C strip: 1 A fragment feeds 4 WMMAs.
// Launch with blockDim=64 and exactly (M/16)*(N/64) waves (EXEC all-ones).
// ---------------------------------------------------------------------------
__global__ void gat_wmma_gemm(const __bf16* __restrict__ A,
                              const __bf16* __restrict__ Bp,
                              float* __restrict__ C,
                              int M, int N, int K) {
  int wave = (int)((blockIdx.x * blockDim.x + threadIdx.x) >> 5);
  int lane = threadIdx.x & 31;
  int NG     = N >> 6;   // groups of 4 N-tiles
  int tilesN = N >> 4;
  int tm = wave / NG;
  int tg = wave % NG;
  int m0 = tm << 4;
  int hl = lane >> 4, l = lane & 15;

  v8f acc0 = {}, acc1 = {}, acc2 = {}, acc3 = {};
  const v8bf*  Arow  = (const v8bf*)(A + (size_t)(m0 + l) * K);
  const v16bf* Bfrag = (const v16bf*)Bp;

  for (int k0 = 0; k0 < K; k0 += 32) {
    // A 16x32 fragment: lane half hl holds K = k0+hl*16..+15.
    v8bf alo = Arow[(k0 + hl * 16) >> 3];
    v8bf ahi = Arow[((k0 + hl * 16) >> 3) + 1];
    v16bf av = __builtin_shufflevector(alo, ahi, 0, 1, 2, 3, 4, 5, 6, 7,
                                       8, 9, 10, 11, 12, 13, 14, 15);
    size_t base = ((size_t)(k0 >> 5) * tilesN + (tg << 2)) * 32 + lane;
    v16bf b0 = Bfrag[base];
    v16bf b1 = Bfrag[base + 32];
    v16bf b2 = Bfrag[base + 64];
    v16bf b3 = Bfrag[base + 96];
    acc0 = __builtin_amdgcn_wmma_f32_16x16x32_bf16(false, av, false, b0, (short)0, acc0, false, false);
    acc1 = __builtin_amdgcn_wmma_f32_16x16x32_bf16(false, av, false, b1, (short)0, acc1, false, false);
    acc2 = __builtin_amdgcn_wmma_f32_16x16x32_bf16(false, av, false, b2, (short)0, acc2, false, false);
    acc3 = __builtin_amdgcn_wmma_f32_16x16x32_bf16(false, av, false, b3, (short)0, acc3, false, false);
  }

  // C 16x16 f32 layout: VGPR r -> M = m0 + hl*8 + r, N-col = tile_n0 + l.
  float* Cbase = C + (size_t)(m0 + hl * 8) * N + (tg << 6) + l;
#pragma unroll
  for (int r = 0; r < 8; ++r) {
    float* Crow = Cbase + (size_t)r * N;
    Crow[0]  = acc0[r];
    Crow[16] = acc1[r];
    Crow[32] = acc2[r];
    Crow[48] = acc3[r];
  }
}

// ---------------------------------------------------------------------------
// Per-(node,head) attention dot products: el = <feat, al>, er = <feat, ar>.
// ---------------------------------------------------------------------------
__global__ void gat_attn_scores(const float* __restrict__ feat,
                                const float* __restrict__ al,
                                const float* __restrict__ ar,
                                float* __restrict__ el, float* __restrict__ er,
                                int N, int H, int D) {
  int i = blockIdx.x * blockDim.x + threadIdx.x;
  if (i >= N * H) return;
  int n = i / H, h = i % H;
  const float* f = feat + ((size_t)n * H + h) * D;
  const float* L = al + (size_t)h * D;
  const float* R = ar + (size_t)h * D;
  float sl = 0.f, sr = 0.f;
  for (int d = 0; d < D; ++d) {
    float v = f[d];
    sl += v * L[d];
    sr += v * R[d];
  }
  el[i] = sl;
  er[i] = sr;
}

__device__ __forceinline__ void atomic_max_float_bits(int* addr, float val) {
  // Correct for mixed signs when cell initialized to 0xFFFFFFFF.
  if (val >= 0.0f) atomicMax(addr, __float_as_int(val));
  else             atomicMin((unsigned int*)addr, __float_as_uint(val));
}

// Edge pass 1: e = leaky_relu(el[src]+er[dst]); segment max over dst.
__global__ void gat_edge_max(const int* __restrict__ src, const int* __restrict__ dst,
                             const float* __restrict__ el, const float* __restrict__ er,
                             int* __restrict__ emax, int E, int H) {
  int i = blockIdx.x * blockDim.x + threadIdx.x;
  if (i >= E * H) return;
  int ed = i / H, h = i % H;
  int s = src[ed], d = dst[ed];
  float v = el[s * H + h] + er[d * H + h];
  v = v > 0.f ? v : v * NEG_SLOPE;
  atomic_max_float_bits(&emax[d * H + h], v);
}

// Edge pass 2: ee = exp(e - emax[dst]); segment sum into denom.
__global__ void gat_edge_exp(const int* __restrict__ src, const int* __restrict__ dst,
                             const float* __restrict__ el, const float* __restrict__ er,
                             const int* __restrict__ emax,
                             float* __restrict__ ee, float* __restrict__ denom,
                             int E, int H) {
  int i = blockIdx.x * blockDim.x + threadIdx.x;
  if (i >= E * H) return;
  int ed = i / H, h = i % H;
  int s = src[ed], d = dst[ed];
  float v = el[s * H + h] + er[d * H + h];
  v = v > 0.f ? v : v * NEG_SLOPE;
  float ex = expf(v - __int_as_float(emax[d * H + h]));
  ee[i] = ex;
  atomicAdd(&denom[d * H + h], ex);
}

// Edge pass 3: out[dst] += feat[src] * (ee/denom[dst]); one thread per
// (edge, float4 chunk). 128-bit gather + 4 f32 atomic adds (L2-resident).
__global__ void gat_edge_scatter(const int* __restrict__ src, const int* __restrict__ dst,
                                 const float* __restrict__ feat,
                                 const float* __restrict__ ee,
                                 const float* __restrict__ denom,
                                 float* __restrict__ out,
                                 int E, int H, int D) {
  int C = (H * D) >> 2;
  long long i = (long long)blockIdx.x * blockDim.x + threadIdx.x;
  if (i >= (long long)E * C) return;
  int ed = (int)(i / C);
  int c  = (int)(i % C);
  int h  = (c << 2) / D;
  int s = src[ed], d = dst[ed];
  float alpha = ee[(size_t)ed * H + h] / denom[d * H + h];
  float4 f = ((const float4*)(feat + (size_t)s * H * D))[c];
  float* o = out + (size_t)d * H * D + ((size_t)c << 2);
  atomicAdd(o + 0, f.x * alpha);
  atomicAdd(o + 1, f.y * alpha);
  atomicAdd(o + 2, f.z * alpha);
  atomicAdd(o + 3, f.w * alpha);
}

// Layer-0 epilogue: h = elu(acc + bias) written as bf16 (feeds layer-1 WMMA).
__global__ void gat_bias_elu_bf16(const float* __restrict__ acc,
                                  const float* __restrict__ bias,
                                  __bf16* __restrict__ out, int N, int F) {
  int i = blockIdx.x * blockDim.x + threadIdx.x;
  if (i >= N * F) return;
  float v = acc[i] + bias[i % F];
  v = v > 0.f ? v : expm1f(v);
  out[i] = (__bf16)v;
}

// Final epilogue: out = elu(out + bias), f32.
__global__ void gat_bias_elu_f32(float* __restrict__ io, const float* __restrict__ bias,
                                 int N, int F) {
  int i = blockIdx.x * blockDim.x + threadIdx.x;
  if (i >= N * F) return;
  float v = io[i] + bias[i % F];
  io[i] = v > 0.f ? v : expm1f(v);
}

extern "C" void kernel_launch(void* const* d_in, const int* in_sizes, int n_in,
                              void* d_out, int out_size, void* d_ws, size_t ws_size,
                              hipStream_t stream) {
  const float* x   = (const float*)d_in[0];
  const int*   src = (const int*)d_in[1];
  const int*   dst = (const int*)d_in[2];
  const float* W0  = (const float*)d_in[3];
  const float* al0 = (const float*)d_in[4];
  const float* ar0 = (const float*)d_in[5];
  const float* b0  = (const float*)d_in[6];
  const float* W1  = (const float*)d_in[7];
  const float* al1 = (const float*)d_in[8];
  const float* ar1 = (const float*)d_in[9];
  const float* b1  = (const float*)d_in[10];
  float* out = (float*)d_out;

  const int N = GAT_N, E = GAT_E;

  // ---- Workspace layout (byte offsets, 64B-aligned chunks) ----
  char* p = (char*)d_ws;
  float*  feat = (float*)p;  p += (size_t)N * 256 * 4;   // GEMM out (layer1: N*128)
  float*  acc0 = (float*)p;  p += (size_t)N * 256 * 4;   // layer-0 scatter target
  __bf16* h0bf = (__bf16*)p; p += (size_t)N * 256 * 2;   // layer-0 output, bf16
  // x_bf16 region is dead after GEMM0; reuse it as the edge 'ee' buffer.
  __bf16* xbf  = (__bf16*)p;
  float*  ee   = (float*)p;  p += (size_t)E * 4 * 4;     // max(N*128*2, E*4*4) bytes
  __bf16* w0p  = (__bf16*)p; p += (size_t)128 * 256 * 2;
  __bf16* w1p  = (__bf16*)p; p += (size_t)256 * 128 * 2;
  float*  el   = (float*)p;  p += (size_t)N * 4 * 4;
  float*  er   = (float*)p;  p += (size_t)N * 4 * 4;
  int*    emax = (int*)p;    p += (size_t)N * 4 * 4;
  float*  denom= (float*)p;  p += (size_t)N * 4 * 4;

  const int T = 256;

  // ---------------- Layer 0: IN=128 -> 4 heads x 64 ----------------
  hipMemsetAsync(acc0, 0, (size_t)N * 256 * sizeof(float), stream);
  hipMemsetAsync(denom, 0, (size_t)N * 4 * sizeof(float), stream);
  hipMemsetAsync(emax, 0xFF, (size_t)N * 4 * sizeof(int), stream);

  gat_cvt_bf16<<<(N * 128 / 4 + T - 1) / T, T, 0, stream>>>(x, xbf, N * 128 / 4);
  gat_pack_w<<<(128 * 256 + T - 1) / T, T, 0, stream>>>(W0, w0p, 128, 256);
  {  // feat = x @ W0 : (N/16)*(256/64) waves, 2 waves/block
    int waves = (N / 16) * (256 / 64);
    gat_wmma_gemm<<<waves / 2, 64, 0, stream>>>(xbf, w0p, feat, N, 256, 128);
  }
  gat_attn_scores<<<(N * 4 + T - 1) / T, T, 0, stream>>>(feat, al0, ar0, el, er, N, 4, 64);
  gat_edge_max<<<(E * 4 + T - 1) / T, T, 0, stream>>>(src, dst, el, er, emax, E, 4);
  gat_edge_exp<<<(E * 4 + T - 1) / T, T, 0, stream>>>(src, dst, el, er, emax, ee, denom, E, 4);
  {
    long long work = (long long)E * 64;  // 64 float4 chunks per edge (H*D=256)
    gat_edge_scatter<<<(unsigned)((work + T - 1) / T), T, 0, stream>>>(
        src, dst, feat, ee, denom, acc0, E, 4, 64);
  }
  gat_bias_elu_bf16<<<(N * 256 + T - 1) / T, T, 0, stream>>>(acc0, b0, h0bf, N, 256);

  // ---------------- Layer 1: IN=256 -> 1 head x 128 ----------------
  hipMemsetAsync(out, 0, (size_t)N * 128 * sizeof(float), stream);
  hipMemsetAsync(denom, 0, (size_t)N * sizeof(float), stream);
  hipMemsetAsync(emax, 0xFF, (size_t)N * sizeof(int), stream);

  gat_pack_w<<<(256 * 128 + T - 1) / T, T, 0, stream>>>(W1, w1p, 256, 128);
  {  // feat = h0 @ W1 : (N/16)*(128/64) waves, 2 waves/block
    int waves = (N / 16) * (128 / 64);
    gat_wmma_gemm<<<waves / 2, 64, 0, stream>>>(h0bf, w1p, feat, N, 128, 256);
  }
  gat_attn_scores<<<(N + T - 1) / T, T, 0, stream>>>(feat, al1, ar1, el, er, N, 1, 128);
  gat_edge_max<<<(E + T - 1) / T, T, 0, stream>>>(src, dst, el, er, emax, E, 1);
  gat_edge_exp<<<(E + T - 1) / T, T, 0, stream>>>(src, dst, el, er, emax, ee, denom, E, 1);
  {
    long long work = (long long)E * 32;  // 32 float4 chunks per edge (H*D=128)
    gat_edge_scatter<<<(unsigned)((work + T - 1) / T), T, 0, stream>>>(
        src, dst, feat, ee, denom, out, E, 1, 128);
  }
  gat_bias_elu_f32<<<(N * 128 + T - 1) / T, T, 0, stream>>>(out, b1, N, 128);
}
